// GoalDecoder_13683765805300
// MI455X (gfx1250) — compile-verified, matrix operand order checked by reading, not run on hardware
//
#include <hip/hip_runtime.h>
#include <hip/hip_bf16.h>
#include <stdint.h>

#define PRED_LEN 12
#define IN_DIM   64
#define HID      256
#define KDIM     320            // 64 (x) + 256 (h)
#define NGATES   1024
#define BATCHSZ  32768
#define MTILE    32             // batch rows per block (2 x 16 WMMA M-tiles)
#define ASTRIDE  328            // padded K stride (ushorts) for A in LDS
#define HFSTRIDE 260            // padded stride for f32 h copy

typedef __attribute__((ext_vector_type(16))) __bf16 v16bf;
typedef __attribute__((ext_vector_type(8)))  float  v8f;

__device__ __forceinline__ unsigned short f2bf(float f) {
    union { float f; uint32_t u; } c; c.f = f;
    uint32_t u = c.u;
    uint32_t r = u + 0x7FFFu + ((u >> 16) & 1u);   // round-to-nearest-even
    return (unsigned short)(r >> 16);
}

__device__ __forceinline__ float fsig(float x) {
    return 1.0f / (1.0f + __expf(-x));
}
__device__ __forceinline__ float ftanh(float x) {
    float ax = fabsf(x);
    float e  = __expf(-2.0f * ax);
    float t  = (1.0f - e) / (1.0f + e);
    return copysignf(t, x);
}

// Pack [W_ih | W_hh] into bf16 (1024 x 320) row-major, and fuse biases.
__global__ __launch_bounds__(256) void prep_kernel(
    const float* __restrict__ W_ih, const float* __restrict__ W_hh,
    const float* __restrict__ b_ih, const float* __restrict__ b_hh,
    unsigned short* __restrict__ Wc, float* __restrict__ bc)
{
    int gid = blockIdx.x * 256 + threadIdx.x;
    if (gid < NGATES * KDIM) {
        int n = gid / KDIM, k = gid % KDIM;
        float v = (k < IN_DIM) ? W_ih[n * IN_DIM + k] : W_hh[n * HID + (k - IN_DIM)];
        Wc[gid] = f2bf(v);
    }
    if (gid < NGATES) bc[gid] = b_ih[gid] + b_hh[gid];
}

__global__ __launch_bounds__(256) void lstm_kernel(
    const float* __restrict__ goal_real,     // (13, B, 2)
    const float* __restrict__ h0g,           // (B, 256)
    const unsigned short* __restrict__ Wc,   // (1024, 320) bf16
    const float* __restrict__ bc,            // (1024) fused bias
    const float* __restrict__ W_hp,          // (2, 256)
    const float* __restrict__ b_hp,          // (2)
    const float* __restrict__ W_emb,         // (64, 2)
    const float* __restrict__ b_emb,         // (64)
    float* __restrict__ out)                 // (12, B, 2)
{
    __shared__ unsigned short Alds[MTILE * ASTRIDE];   // bf16 [x | h], 32 x 320
    __shared__ float          Hf[MTILE * HFSTRIDE];    // f32 h_new for projection

    const int tid  = threadIdx.x;
    const int wave = tid >> 5;
    const int lane = tid & 31;
    const int l16  = lane & 15;
    const int lhi  = lane >> 4;          // 0/1: K chunk base 0/8 (+16)
    const int cb   = wave * 32;          // this wave's hidden-unit column base
    const int b0   = blockIdx.x * MTILE;

    // ---- init h0 (bf16) into A columns 64..319 ----
    {
        int col = tid;                   // 0..255
        #pragma unroll 4
        for (int r = 0; r < MTILE; ++r)
            Alds[r * ASTRIDE + IN_DIM + col] = f2bf(h0g[(size_t)(b0 + r) * HID + col]);
    }
    // ---- per-thread embedding constants + x_0 into A columns 0..63 ----
    const int   ed  = tid & 63;
    const int   er4 = tid >> 6;          // 0..3 -> rows er4 + 4*rr, rr=0..7
    const float we0 = W_emb[ed * 2 + 0];
    const float we1 = W_emb[ed * 2 + 1];
    const float beb = b_emb[ed];
    #pragma unroll
    for (int rr = 0; rr < 8; ++rr) {
        int r = er4 + 4 * rr;
        const float* gr = goal_real + (size_t)1 * BATCHSZ * 2 + (size_t)(b0 + r) * 2;
        Alds[r * ASTRIDE + ed] = f2bf(gr[0] * we0 + gr[1] * we1 + beb);
    }

    // ---- preload fused gate biases (constant over steps) ----
    float bias_[4][2];
    #pragma unroll
    for (int g = 0; g < 4; ++g)
        #pragma unroll
        for (int t = 0; t < 2; ++t)
            bias_[g][t] = bc[g * 256 + cb + t * 16 + l16];

    // ---- preload projection weights ----
    float whp[2][8];
    #pragma unroll
    for (int oi = 0; oi < 2; ++oi)
        #pragma unroll
        for (int k = 0; k < 8; ++k)
            whp[oi][k] = W_hp[oi * HID + lane + 32 * k];
    const float bhp0 = b_hp[0], bhp1 = b_hp[1];

    // ---- cell state in registers: [t][m], rows m*16 + r + 8*lhi, cols cb+16t+l16
    v8f cst[2][2];
    #pragma unroll
    for (int t = 0; t < 2; ++t)
        #pragma unroll
        for (int m = 0; m < 2; ++m) cst[t][m] = (v8f){0,0,0,0,0,0,0,0};

    __syncthreads();

    for (int step = 0; step < PRED_LEN; ++step) {
        v8f acc[4][2][2];
        #pragma unroll
        for (int g = 0; g < 4; ++g)
            #pragma unroll
            for (int t = 0; t < 2; ++t)
                #pragma unroll
                for (int m = 0; m < 2; ++m) acc[g][t][m] = (v8f){0,0,0,0,0,0,0,0};

        // ---- GEMM: gates(32 x 1024-slice) += A(32x320) @ W^T, bf16 WMMA ----
        // Each B fragment (one global_load pair) is reused by both M-tiles.
        #pragma unroll 1
        for (int kk = 0; kk < KDIM / 32; ++kk) {
            union { uint4 u[2]; v16bf v; } afr[2];
            #pragma unroll
            for (int m = 0; m < 2; ++m) {
                const unsigned short* ab =
                    Alds + (m * 16 + l16) * ASTRIDE + kk * 32 + lhi * 8;
                afr[m].u[0] = *(const uint4*)(ab);
                afr[m].u[1] = *(const uint4*)(ab + 16);
            }
            #pragma unroll
            for (int g = 0; g < 4; ++g) {
                #pragma unroll
                for (int t = 0; t < 2; ++t) {
                    union { uint4 u[2]; v16bf v; } bfr;
                    const unsigned short* bb =
                        Wc + (size_t)(g * 256 + cb + t * 16 + l16) * KDIM + kk * 32 + lhi * 8;
                    bfr.u[0] = *(const uint4*)(bb);
                    bfr.u[1] = *(const uint4*)(bb + 16);
                    #pragma unroll
                    for (int m = 0; m < 2; ++m)
                        acc[g][t][m] = __builtin_amdgcn_wmma_f32_16x16x32_bf16(
                            false, afr[m].v, false, bfr.v, (short)0, acc[g][t][m],
                            false, false);
                }
            }
        }

        // ---- LSTM elementwise (all four gates are wave-local) ----
        float hvals[2][2][8];
        #pragma unroll
        for (int t = 0; t < 2; ++t) {
            #pragma unroll
            for (int m = 0; m < 2; ++m) {
                #pragma unroll
                for (int r = 0; r < 8; ++r) {
                    float iv = fsig (acc[0][t][m][r] + bias_[0][t]);
                    float fv = fsig (acc[1][t][m][r] + bias_[1][t]);
                    float gv = ftanh(acc[2][t][m][r] + bias_[2][t]);
                    float ov = fsig (acc[3][t][m][r] + bias_[3][t]);
                    float cn = fv * cst[t][m][r] + iv * gv;
                    cst[t][m][r]   = cn;
                    hvals[t][m][r] = ov * ftanh(cn);
                }
            }
        }

        __syncthreads();   // all waves finished reading A for this step

        // ---- write h_new back into A (bf16) + f32 copy for projection ----
        #pragma unroll
        for (int t = 0; t < 2; ++t) {
            int col = cb + t * 16 + l16;
            #pragma unroll
            for (int m = 0; m < 2; ++m) {
                #pragma unroll
                for (int r = 0; r < 8; ++r) {
                    int row = m * 16 + r + lhi * 8;
                    Alds[row * ASTRIDE + IN_DIM + col] = f2bf(hvals[t][m][r]);
                    Hf[row * HFSTRIDE + col]           = hvals[t][m][r];
                }
            }
        }
        // ---- write x_{t+1} (teacher forcing: from goal_real embedding) ----
        if (step + 1 < PRED_LEN) {
            #pragma unroll
            for (int rr = 0; rr < 8; ++rr) {
                int r = er4 + 4 * rr;
                const float* gr = goal_real + (size_t)(step + 2) * BATCHSZ * 2
                                            + (size_t)(b0 + r) * 2;
                Alds[r * ASTRIDE + ed] = f2bf(gr[0] * we0 + gr[1] * we1 + beb);
            }
        }

        __syncthreads();   // h_new / x_{t+1} visible; Hf ready

        // ---- projection: wave w handles rows 4w .. 4w+3 ----
        #pragma unroll
        for (int rr = 0; rr < 4; ++rr) {
            int row = wave * 4 + rr;
            #pragma unroll
            for (int oi = 0; oi < 2; ++oi) {
                float s = 0.f;
                #pragma unroll
                for (int k = 0; k < 8; ++k)
                    s += Hf[row * HFSTRIDE + lane + 32 * k] * whp[oi][k];
                #pragma unroll
                for (int off = 16; off > 0; off >>= 1)
                    s += __shfl_down(s, off, 32);
                if (lane == 0)
                    out[(size_t)step * BATCHSZ * 2 + (size_t)(b0 + row) * 2 + oi] =
                        s + (oi ? bhp1 : bhp0);
            }
        }
    }
}

extern "C" void kernel_launch(void* const* d_in, const int* in_sizes, int n_in,
                              void* d_out, int out_size, void* d_ws, size_t ws_size,
                              hipStream_t stream) {
    const float* goal_real = (const float*)d_in[0];
    const float* h0        = (const float*)d_in[1];
    const float* W_ih      = (const float*)d_in[2];
    const float* W_hh      = (const float*)d_in[3];
    const float* b_ih      = (const float*)d_in[4];
    const float* b_hh      = (const float*)d_in[5];
    const float* W_hp      = (const float*)d_in[6];
    const float* b_hp      = (const float*)d_in[7];
    const float* W_emb     = (const float*)d_in[8];
    const float* b_emb     = (const float*)d_in[9];
    float* out = (float*)d_out;

    unsigned short* Wc = (unsigned short*)d_ws;                                  // 640 KB
    float* bc = (float*)((char*)d_ws + (size_t)NGATES * KDIM * sizeof(unsigned short)); // +4 KB

    prep_kernel<<<(NGATES * KDIM + 255) / 256, 256, 0, stream>>>(
        W_ih, W_hh, b_ih, b_hh, Wc, bc);

    lstm_kernel<<<BATCHSZ / MTILE, 256, 0, stream>>>(
        goal_real, h0, Wc, bc, W_hp, b_hp, W_emb, b_emb, out);
}
